// GraphMultiHeadAttentionLayer_12541304504947
// MI455X (gfx1250) — matrix operand order repaired and checked by reading, not compiled, not run on hardware
//
#include <hip/hip_runtime.h>
#include <hip/hip_bf16.h>
#include <cstdint>
#include <cmath>

// Problem sizes (fixed by the reference)
#define BB 4
#define NN 2048
#define FF 128
#define DD 64
#define HH 4

typedef __attribute__((ext_vector_type(16))) _Float16 v16h;
typedef __attribute__((ext_vector_type(8)))  _Float16 v8h;
typedef __attribute__((ext_vector_type(8)))  float    v8f;
typedef __attribute__((ext_vector_type(4)))  uint32_t u32x4;
typedef __attribute__((ext_vector_type(8)))  uint32_t u32x8;

// ---- WMMA fragment loads (CDNA5 16-bit layouts, wave32) -------------------
// A 16x32 (MxK), row-major LDS tile with row stride `stride` (halves).
// Operand element e: e<8 -> K = half*8 + e ; e>=8 -> K = 16 + half*8 + (e-8).
// Both 8-element runs are contiguous -> two ds_load_b128 per lane.
__device__ __forceinline__ v16h frag_a_rowmajor(const _Float16* tile, int stride) {
  const int lane = threadIdx.x & 31;
  const int m    = lane & 15;
  const int half = lane >> 4;
  const _Float16* p = tile + m * stride + half * 8;
  v16h r;
  *((v8h*)&r)     = *(const v8h*)(p);
  *((v8h*)&r + 1) = *(const v8h*)(p + 16);
  return r;
}

// B 32x16 (KxN) from a column-major tile Bt[col][k] with row stride `stride`.
// Operand element e: e<8 -> K = half*8 + e ; e>=8 -> K = 16 + half*8 + (e-8) -> two ds_load_b128 per lane.
__device__ __forceinline__ v16h frag_b_colmajor(const _Float16* Bt, int col0,
                                                int kbase, int stride) {
  const int lane = threadIdx.x & 31;
  const int n    = lane & 15;
  const int half = lane >> 4;
  const _Float16* p = Bt + (col0 + n) * stride + kbase + half * 8;
  v16h r;
  *((v8h*)&r)     = *(const v8h*)(p);
  *((v8h*)&r + 1) = *(const v8h*)(p + 16);
  return r;
}

// ---- Kernel 1: h = X @ W_h (WMMA) + s_i = h_i . a_h -----------------------
// grid: (N/16, H*B), block 128 (wave w owns d-columns 16w..16w+15).
// h is written d-major: hT[hb][d][node] (f16) so phase-3 B tiles are a plain
// 2D TDM tile; each lane emits one 16-byte global_store_b128.
#define XS_STRIDE (FF + 8)    // 136 halves = 272 B (16B-aligned rows)
#define WS_STRIDE (FF + 8)
__global__ __launch_bounds__(128) void gat_proj(
    const float* __restrict__ X, const float* __restrict__ Wk,
    const float* __restrict__ attnk,
    _Float16* __restrict__ hT, float* __restrict__ s) {
  __shared__ _Float16 Xs[16 * XS_STRIDE];        // A tile, row-major
  __shared__ _Float16 WsT[DD * WS_STRIDE];       // W^T: [d][f], column-major B
  __shared__ float    Cs[16 * DD];

  const int hb   = blockIdx.y;          // h*B + b
  const int h    = hb / BB;
  const int b    = hb % BB;
  const int row0 = blockIdx.x * 16;
  const int tid  = threadIdx.x;
  const int wave = tid >> 5;
  const int lane = tid & 31;

  for (int idx = tid; idx < 16 * FF; idx += 128) {
    const int r = idx / FF, f = idx % FF;
    Xs[r * XS_STRIDE + f] = (_Float16)X[((size_t)b * NN + row0 + r) * FF + f];
  }
  for (int idx = tid; idx < FF * DD; idx += 128) {
    const int f = idx / DD, d = idx % DD;
    WsT[d * WS_STRIDE + f] = (_Float16)Wk[(size_t)h * FF * DD + idx];
  }
  __syncthreads();

  v8f acc = {};
#pragma unroll
  for (int kk = 0; kk < FF / 32; ++kk) {
    v16h af = frag_a_rowmajor(&Xs[kk * 32], XS_STRIDE);
    v16h bf = frag_b_colmajor(WsT, wave * 16, kk * 32, WS_STRIDE);
    acc = __builtin_amdgcn_wmma_f32_16x16x32_f16(false, af, false, bf,
                                                 (short)0, acc, false, false);
  }

  const int n    = lane & 15;
  const int half = lane >> 4;
  const int d    = wave * 16 + n;
  v8h hv;
#pragma unroll
  for (int r = 0; r < 8; ++r) {
    Cs[(r + half * 8) * DD + d] = acc[r];   // C/D layout: M = r + 8*half
    hv[r] = (_Float16)acc[r];
  }
  // 8 consecutive nodes of one d-column -> single b128 store
  *(v8h*)&hT[((size_t)hb * DD + d) * NN + row0 + half * 8] = hv;
  __syncthreads();

  if (tid < 16) {
    float acc_s = 0.f;
#pragma unroll
    for (int dd = 0; dd < DD; ++dd)
      acc_s += Cs[tid * DD + dd] * attnk[h * DD + dd];
    s[(size_t)hb * NN + row0 + tid] = acc_s;
  }
}

// ---- Kernel 2: per-(h,b) max over s ---------------------------------------
__global__ __launch_bounds__(256) void gat_smax(const float* __restrict__ s,
                                                float* __restrict__ smax) {
  __shared__ float red[256];
  const int hb = blockIdx.x;
  float m = -1e30f;
  for (int i = threadIdx.x; i < NN; i += 256)
    m = fmaxf(m, s[(size_t)hb * NN + i]);
  red[threadIdx.x] = m;
  __syncthreads();
  for (int off = 128; off > 0; off >>= 1) {
    if (threadIdx.x < (unsigned)off)
      red[threadIdx.x] = fmaxf(red[threadIdx.x], red[threadIdx.x + off]);
    __syncthreads();
  }
  if (threadIdx.x == 0) smax[hb] = red[0];
}

// ---- Kernel 3: masked softmax @ h, flash-style, h tiles via TDM -----------
// e_ij = relu(s_i+s_j) <= m_i := max(0, s_i + smax) so exp(e - m_i) <= 1.
// grid: (N/16, H*B), block 128.
#define PS_STRIDE 40          // 80 B rows: 16B-aligned, de-conflicted
#define HS_STRIDE 40          // produced by TDM pad: 16 DW data + 4 DW pad
__global__ __launch_bounds__(128) void gat_attn(
    const int* __restrict__ Adj, const float* __restrict__ s,
    const float* __restrict__ smax, const _Float16* __restrict__ hT,
    float* __restrict__ hout) {
  __shared__ _Float16 Ps[16 * PS_STRIDE];     // P tile (A operand), row-major
  __shared__ _Float16 HsT[DD * HS_STRIDE];    // h^T tile [d][k] (B operand)
  __shared__ float si[16];
  __shared__ float mi[16];
  __shared__ float den[16];

  const int hb   = blockIdx.y;
  const int row0 = blockIdx.x * 16;
  const int tid  = threadIdx.x;
  const int wave = tid >> 5;
  const int lane = tid & 31;

  if (tid < 16) {
    const float v = s[(size_t)hb * NN + row0 + tid];
    si[tid]  = v;
    mi[tid]  = fmaxf(0.f, v + smax[hb]);
    den[tid] = 0.f;
  }

  // D# group1 is loop-invariant: data_size=2B, pad 4 DWORDs every 16 DWORDs,
  // tensor = [dim0=N, dim1=D] stride N, tile = 32 x 64.
  u32x8 g1;
  g1[0] = (1u << 16) | (1u << 20) | (3u << 22) | (3u << 25);
  g1[1] = (uint32_t)(NN & 0xFFFF) << 16;            // tensor_dim0[15:0]
  g1[2] = (uint32_t)(NN >> 16) | ((DD & 0xFFFF) << 16); // td0[31:16] | td1[15:0]
  g1[3] = (32u << 16);                              // tile_dim0 = 32
  g1[4] = (uint32_t)DD;                             // tile_dim1 = 64
  g1[5] = (uint32_t)NN;                             // td0_stride[31:0]
  g1[6] = 0u;
  g1[7] = 0u;
  const uint32_t lds_off = (uint32_t)(uintptr_t)(void*)HsT;
  const _Float16* tile_base = hT + (size_t)hb * DD * NN;

  v8f acc = {};

  for (int m0 = 0; m0 < NN; m0 += 32) {
    __syncthreads();                 // prior chunk's fragment reads complete
    if (wave == 0) {                 // one TDM issue per block per chunk
      const uint64_t gaddr = (uint64_t)(uintptr_t)(tile_base + m0);
      u32x4 g0;
      g0[0] = 1u;                                        // count = 1 (valid)
      g0[1] = lds_off;                                   // lds_addr
      g0[2] = (uint32_t)gaddr;                           // global_addr[31:0]
      g0[3] = (uint32_t)((gaddr >> 32) & 0x01FFFFFFu) |  // global_addr[56:32]
              (2u << 30);                                // type = 2 (image)
      asm volatile("tensor_load_to_lds %0, %1" :: "s"(g0), "s"(g1) : "memory");
    }

    // P tile while the TDM streams: 512 elems / 128 thr = 4 each (same row)
    {
      const int base = tid * 4;
      const int i    = base >> 5;
      const int j0   = base & 31;
      float part = 0.f;
#pragma unroll
      for (int jj = 0; jj < 4; ++jj) {
        const int j = j0 + jj;
        float p = 0.f;
        if (Adj[(size_t)(row0 + i) * NN + m0 + j] > 0) {
          const float sjv = s[(size_t)hb * NN + m0 + j];
          const float e   = fmaxf(0.f, si[i] + sjv);
          p = __expf(e - mi[i]);
        }
        Ps[i * PS_STRIDE + j] = (_Float16)p;
        part += p;
      }
      atomicAdd(&den[i], part);      // ds_add_f32
    }
    if (m0 + 32 < NN)                // warm L2 for the next TDM tile
      __builtin_prefetch(tile_base + (size_t)(tid & 63) * NN + m0 + 32, 0, 0);

    if (wave == 0) __builtin_amdgcn_s_wait_tensorcnt(0);
    __syncthreads();                 // HsT + Ps visible to all waves

    v16h af = frag_a_rowmajor(Ps, PS_STRIDE);
    v16h bf = frag_b_colmajor(HsT, wave * 16, 0, HS_STRIDE);
    acc = __builtin_amdgcn_wmma_f32_16x16x32_f16(false, af, false, bf,
                                                 (short)0, acc, false, false);
  }
  __syncthreads();                   // den fully accumulated

  const int n    = lane & 15;
  const int half = lane >> 4;
#pragma unroll
  for (int r = 0; r < 8; ++r) {
    const int m = r + half * 8;
    const float dsum = fmaxf(den[m], 1e-20f);
    hout[((size_t)hb * NN + row0 + m) * DD + wave * 16 + n] = acc[r] / dsum;
  }
}

// ---- Kernel 4: mean over heads + leaky_relu(0.2) --------------------------
__global__ __launch_bounds__(256) void gat_final(const float* __restrict__ hout,
                                                 float* __restrict__ out) {
  const size_t idx   = (size_t)blockIdx.x * 256 + threadIdx.x;
  const size_t total = (size_t)BB * NN * DD;
  if (idx >= total) return;
  float v = 0.f;
#pragma unroll
  for (int h = 0; h < HH; ++h)
    v += hout[(size_t)h * BB * NN * DD + idx];
  v *= (1.0f / HH);
  out[idx] = v > 0.f ? v : 0.2f * v;
}

extern "C" void kernel_launch(void* const* d_in, const int* in_sizes, int n_in,
                              void* d_out, int out_size, void* d_ws, size_t ws_size,
                              hipStream_t stream) {
  const float* X  = (const float*)d_in[0];   // [B,N,F] f32
  const int*   A  = (const int*)d_in[1];     // [N,N] i32
  const float* Wk = (const float*)d_in[2];   // [H,F,D] f32
  const float* ak = (const float*)d_in[3];   // [H,D] f32
  float* out = (float*)d_out;                // [B,N,D] f32

  char* ws = (char*)d_ws;
  size_t off = 0;
  _Float16* hT = (_Float16*)(ws + off);               // [H*B][D][N] f16, 4 MiB
  off += (size_t)HH * BB * DD * NN * sizeof(_Float16);
  float* s = (float*)(ws + off);                      // [H*B][N] f32
  off += (size_t)HH * BB * NN * sizeof(float);
  float* smax = (float*)(ws + off);                   // [H*B] f32 (padded)
  off += 256;
  float* hout = (float*)(ws + off);                   // [H*B][N][D] f32, 8 MiB
  (void)in_sizes; (void)n_in; (void)out_size; (void)ws_size;

  gat_proj<<<dim3(NN / 16, HH * BB), 128, 0, stream>>>(X, Wk, ak, hT, s);
  gat_smax<<<dim3(HH * BB), 256, 0, stream>>>(s, smax);
  gat_attn<<<dim3(NN / 16, HH * BB), 128, 0, stream>>>(A, s, smax, hT, hout);
  const size_t total = (size_t)BB * NN * DD;
  gat_final<<<dim3((unsigned)((total + 255) / 256)), 256, 0, stream>>>(hout, out);
}
